// XEyTransformerLayer_42468636623144
// MI455X (gfx1250) — compile-verified, matrix operand order checked by reading, not run on hardware
//
#include <hip/hip_runtime.h>
#include <hip/hip_bf16.h>
#include <string.h>

// ---------------------------------------------------------------------------
// Graph-transformer layer (DiGress XEyTransformerLayer) for MI455X / gfx1250.
// Strategy: never materialize (M x DX) edge tensors in HBM. All dense GEMMs
// run through a wave32 WMMA kernel (v_wmma_f32_16x16x32_f16); the edge path
// fuses E1/E2 WMMA + q/k gathers + FiLM + e_out WMMA, with a 2-pass
// recompute for the per-feature segment softmax (atomics for max/sum).
// Staging uses clamped rows + float4 -> half4 vector moves (b128 loads,
// b64 LDS stores) so the tile feeders run at full memory throughput.
// ---------------------------------------------------------------------------

static constexpr int NN   = 20000;
static constexpr int MEDG = 500000;
static constexpr int BG   = 64;
static constexpr int DXc  = 256;
static constexpr int DEc  = 64;
static constexpr int DYc  = 64;
static constexpr int FFXc = 2048;
static constexpr int FFEc = 128;
static constexpr int FFYc = 2048;

static constexpr int LDP  = 40;   // LDS tile row stride in halves (80B: 16B-aligned rows)

typedef __attribute__((ext_vector_type(16))) _Float16 v16h;
typedef __attribute__((ext_vector_type(4)))  _Float16 h4;
typedef __attribute__((ext_vector_type(8)))  float    v8f;

// ---- monotonic float <-> uint mapping for atomic min/max ------------------
__device__ __forceinline__ unsigned flipf(float f) {
  unsigned u = __float_as_uint(f);
  return (u & 0x80000000u) ? ~u : (u | 0x80000000u);
}
__device__ __forceinline__ float unflipf(unsigned u) {
  u = (u & 0x80000000u) ? (u & 0x7FFFFFFFu) : ~u;
  return __uint_as_float(u);
}
static unsigned host_flip(float f) {
  unsigned u; memcpy(&u, &f, 4);
  return (u & 0x80000000u) ? ~u : (u | 0x80000000u);
}

__device__ __forceinline__ h4 cvt_h4(float4 v) {
  h4 h;
  h.x = (_Float16)v.x; h.y = (_Float16)v.y;
  h.z = (_Float16)v.z; h.w = (_Float16)v.w;
  return h;
}

// ---- WMMA fragment K mapping (16-bit A matrix 16x32, ISA 7.12.2) ----------
// lanes 0-15 : K = 0..7 (halves 0-7), 16..23 (halves 8-15)
// lanes 16-31: K = 8..15,             24..31
__device__ __forceinline__ int frag_k(int i, int lane) {
  int kbase = (lane & 16) ? 8 : 0;
  return (i < 8) ? (kbase + i) : (kbase + 8 + i);
}

__device__ __forceinline__ v8f wmma_f16(v16h a, v16h b, v8f c) {
  return __builtin_amdgcn_wmma_f32_16x16x32_f16(false, a, false, b, (short)0, c,
                                                false, false);
}

// ---------------------------------------------------------------------------
// Generic WMMA GEMM: C[M x Nout] = (ACC ? C : 0) + relu?(A[M x K] @ W[Nout x K]^T + bias)
// Block: 256 threads, 64x64 output tile, 8 waves each owning a 16x32 sub-tile.
// Requires: Nout % 64 == 0, K % 32 == 0 (true for all calls here).
// Out-of-range A rows are CLAMPED (not zeroed): row r of A only feeds row r
// of C, and those rows are never stored, so duplicated data is harmless and
// the staging loop stays branch-free and vectorized.
// ---------------------------------------------------------------------------
template<bool RELU, bool ACC>
__global__ __launch_bounds__(256) void gemm_wmma(
    const float* __restrict__ A, const float* __restrict__ W,
    const float* __restrict__ bias, float* __restrict__ C,
    int Mrows, int K, int Nout) {
  __shared__ _Float16 As[64][LDP];
  __shared__ _Float16 Bs[64][LDP];
  const int tid  = threadIdx.x;
  const int lane = tid & 31;
  const int wv   = tid >> 5;
  const int wr   = wv & 3;          // 16-row group
  const int wc   = wv >> 2;         // 32-col group
  const int m0   = blockIdx.y * 64;
  const int n0   = blockIdx.x * 64;
  const int ml   = lane & 15;
  const int rsel = (lane >> 4) << 3;

  v8f acc0 = {0.f,0.f,0.f,0.f,0.f,0.f,0.f,0.f};
  v8f acc1 = acc0;

  const int ksteps = K >> 5;
  for (int kt = 0; kt < ksteps; ++kt) {
    const int k0 = kt << 5;
    __syncthreads();
    // 64x32 tile = 512 float4 per matrix; 2 vector iterations per thread.
#pragma unroll
    for (int i = tid; i < 64 * 8; i += 256) {
      int r  = i >> 3;
      int c4 = (i & 7) << 2;
      int m  = m0 + r;
      if (m >= Mrows) m = Mrows - 1;             // clamp: dead rows, never stored
      float4 av = *(const float4*)(A + (size_t)m * K + k0 + c4);
      float4 bv = *(const float4*)(W + (size_t)(n0 + r) * K + k0 + c4);
      *(h4*)(&As[r][c4]) = cvt_h4(av);
      *(h4*)(&Bs[r][c4]) = cvt_h4(bv);
    }
    if (kt + 1 < ksteps) {  // gfx1250 global_prefetch_b8 for the next A tile
      __builtin_prefetch(&A[(size_t)(m0 + (tid & 63)) * K + k0 + 32], 0, 1);
    }
    __syncthreads();
    v16h a, b0, b1;
#pragma unroll
    for (int i = 0; i < 16; ++i) {
      int kk = frag_k(i, lane);
      a[i]  = As[wr * 16 + ml][kk];
      b0[i] = Bs[wc * 32 + ml][kk];
      b1[i] = Bs[wc * 32 + 16 + ml][kk];
    }
    acc0 = wmma_f16(a, b0, acc0);
    acc1 = wmma_f16(a, b1, acc1);
  }

  for (int r = 0; r < 8; ++r) {
    int m = m0 + wr * 16 + rsel + r;
    if (m >= Mrows) continue;
    int n = n0 + wc * 32 + ml;
    float v0 = acc0[r] + bias[n];
    float v1 = acc1[r] + bias[n + 16];
    if (RELU) { v0 = fmaxf(v0, 0.f); v1 = fmaxf(v1, 0.f); }
    size_t o = (size_t)m * Nout + n;
    if (ACC) { v0 += C[o]; v1 += C[o + 16]; }
    C[o] = v0;
    C[o + 16] = v1;
  }
}

// ---------------------------------------------------------------------------
// Fused edge kernel. One block = 16 edges.
//  PHASE 0: E1/E2 = WMMA(edge_attr, e_mul/e_add); Y = att*(E1+1)+E2;
//           atomicMax of Y into mx[dst,f]; newE = FiLM_y(Y);
//           new_edge = WMMA(newE, e_out) -> ws.newedge.
//  PHASE 1: recompute Y; ex = exp(Y - mx[dst,f]);
//           den[dst,f] += ex;  num[dst,f] += ex * v[src,f].
// ---------------------------------------------------------------------------
struct EdgeSmemGemm { _Float16 w1[256][LDP]; _Float16 w2[256][LDP]; };
struct EdgeSmemOut  { _Float16 wo[64][LDP]; };
union  EdgeSmemU    { EdgeSmemGemm g; EdgeSmemOut o; };

template<int PHASE>
__global__ __launch_bounds__(256) void edge_kernel(
    const int* __restrict__ eidx, const float* __restrict__ eattr,
    const int* __restrict__ batchv,
    const float* __restrict__ qm, const float* __restrict__ km,
    const float* __restrict__ vm,
    const float* __restrict__ emul_w, const float* __restrict__ emul_b,
    const float* __restrict__ eadd_w, const float* __restrict__ eadd_b,
    const float* __restrict__ eout_w, const float* __restrict__ eout_b,
    const float* __restrict__ yea, const float* __restrict__ yem,
    unsigned* __restrict__ mxbuf, float* __restrict__ den,
    float* __restrict__ num, float* __restrict__ newedge) {
  __shared__ EdgeSmemU sm;
  __shared__ _Float16 ea[16][72];
  __shared__ _Float16 neh[16][258];
  __shared__ int sidx[16], didx[16], bidx[16];

  const int tid  = threadIdx.x;
  const int lane = tid & 31;
  const int wv   = tid >> 5;
  const int ml   = lane & 15;
  const int rsel = (lane >> 4) << 3;
  const int e0   = blockIdx.x * 16;

  if (tid < 16) {
    int e  = e0 + tid;
    int ee = (e < MEDG) ? e : (MEDG - 1);
    int s  = eidx[ee];
    sidx[tid] = s;
    didx[tid] = eidx[MEDG + ee];
    bidx[tid] = batchv[s];
  }
  // edge_attr tile 16x64 = 256 float4; 1 vector iteration per thread.
  if (tid < 256) {
    int r  = tid >> 4;
    int c4 = (tid & 15) << 2;
    int e  = e0 + r;
    if (e >= MEDG) e = MEDG - 1;                 // clamp: pad rows never stored
    float4 av = *(const float4*)(eattr + (size_t)e * DEc + c4);
    *(h4*)(&ea[r][c4]) = cvt_h4(av);
  }

  // E1/E2: 16 x 256 outputs, each wave owns a 32-column strip (2 subtiles each).
  v8f a10 = {0.f,0.f,0.f,0.f,0.f,0.f,0.f,0.f};
  v8f a11 = a10, a20 = a10, a21 = a10;
  for (int kt = 0; kt < 2; ++kt) {
    int k0 = kt << 5;
    __syncthreads();
    // 256x32 slice = 2048 float4 per weight matrix; 8 vector iters per thread.
#pragma unroll
    for (int i = tid; i < 256 * 8; i += 256) {
      int n  = i >> 3;
      int c4 = (i & 7) << 2;
      float4 w1v = *(const float4*)(emul_w + (size_t)n * DEc + k0 + c4);
      float4 w2v = *(const float4*)(eadd_w + (size_t)n * DEc + k0 + c4);
      *(h4*)(&sm.g.w1[n][c4]) = cvt_h4(w1v);
      *(h4*)(&sm.g.w2[n][c4]) = cvt_h4(w2v);
    }
    __syncthreads();
    v16h a, b10, b11, b20, b21;
#pragma unroll
    for (int i = 0; i < 16; ++i) {
      int kk = frag_k(i, lane);
      a[i]   = ea[ml][k0 + kk];
      b10[i] = sm.g.w1[wv * 32 + ml][kk];
      b11[i] = sm.g.w1[wv * 32 + 16 + ml][kk];
      b20[i] = sm.g.w2[wv * 32 + ml][kk];
      b21[i] = sm.g.w2[wv * 32 + 16 + ml][kk];
    }
    a10 = wmma_f16(a, b10, a10);
    a11 = wmma_f16(a, b11, a11);
    a20 = wmma_f16(a, b20, a20);
    a21 = wmma_f16(a, b21, a21);
  }

  const float isd = 0.17677669529663687f;  // 1/sqrt(DF=32)
  for (int sub = 0; sub < 2; ++sub) {
    int f = wv * 32 + sub * 16 + ml;
    float e1b = emul_b[f], e2b = eadd_b[f];
    for (int r = 0; r < 8; ++r) {
      int er = rsel + r;
      int e  = e0 + er;
      float Yv = 0.f;
      if (e < MEDG) {
        int s = sidx[er], d = didx[er];
        float att = qm[(size_t)d * DXc + f] * km[(size_t)s * DXc + f] * isd;
        float E1 = (sub == 0 ? a10[r] : a11[r]) + e1b;
        float E2 = (sub == 0 ? a20[r] : a21[r]) + e2b;
        Yv = att * (E1 + 1.f) + E2;
        if (PHASE == 0) {
          atomicMax(&mxbuf[(size_t)d * DXc + f], flipf(Yv));
        } else {
          float mv = unflipf(mxbuf[(size_t)d * DXc + f]);
          float ex = expf(Yv - mv);
          atomicAdd(&den[(size_t)d * DXc + f], ex);
          atomicAdd(&num[(size_t)d * DXc + f], ex * vm[(size_t)s * DXc + f]);
        }
      }
      if (PHASE == 0) {
        float nE = 0.f;
        if (e < MEDG) {
          int b = bidx[er];
          nE = yea[b * DXc + f] + (yem[b * DXc + f] + 1.f) * Yv;
        }
        neh[er][f] = (_Float16)nE;
      }
    }
  }

  if (PHASE == 0) {
    // new_edge = newE(16x256) @ e_out_w(64x256)^T; waves 0..3 own 16-col tiles.
    v8f acco = {0.f,0.f,0.f,0.f,0.f,0.f,0.f,0.f};
    for (int kt = 0; kt < 8; ++kt) {
      int k0 = kt << 5;
      __syncthreads();
      // 64x32 slice = 512 float4; 2 vector iterations per thread.
#pragma unroll
      for (int i = tid; i < 64 * 8; i += 256) {
        int n  = i >> 3;
        int c4 = (i & 7) << 2;
        float4 wv4 = *(const float4*)(eout_w + (size_t)n * DXc + k0 + c4);
        *(h4*)(&sm.o.wo[n][c4]) = cvt_h4(wv4);
      }
      __syncthreads();
      if (wv < 4) {
        v16h a, b;
#pragma unroll
        for (int i = 0; i < 16; ++i) {
          int kk = frag_k(i, lane);
          a[i] = neh[ml][k0 + kk];
          b[i] = sm.o.wo[wv * 16 + ml][kk];
        }
        acco = wmma_f16(a, b, acco);
      }
    }
    if (wv < 4) {
      for (int r = 0; r < 8; ++r) {
        int e = e0 + rsel + r;
        if (e < MEDG) {
          int n = wv * 16 + ml;
          newedge[(size_t)e * DEc + n] = acco[r] + eout_b[n];
        }
      }
    }
  }
}

// ---------------------------------------------------------------------------
// Elementwise / reduction helper kernels
// ---------------------------------------------------------------------------
__global__ void fill_f32(float* p, float v, int n) {
  int i = blockIdx.x * 256 + threadIdx.x;
  if (i < n) p[i] = v;
}
__global__ void fill_u32(unsigned* p, unsigned v, int n) {
  int i = blockIdx.x * 256 + threadIdx.x;
  if (i < n) p[i] = v;
}

// out[row] = LN(a[row] + add[row]) * g + beta ; blockDim == D (64 or 256)
__global__ void ln_kernel(const float* __restrict__ a, const float* __restrict__ add,
                          const float* __restrict__ g, const float* __restrict__ beta,
                          float* __restrict__ out, int D) {
  __shared__ float red[256];
  int row = blockIdx.x, t = threadIdx.x;
  size_t o = (size_t)row * D + t;
  float x = a[o] + add[o];
  red[t] = x;
  __syncthreads();
  for (int s = D >> 1; s > 0; s >>= 1) { if (t < s) red[t] += red[t + s]; __syncthreads(); }
  float mean = red[0] / (float)D;
  __syncthreads();
  float d0 = x - mean;
  red[t] = d0 * d0;
  __syncthreads();
  for (int s = D >> 1; s > 0; s >>= 1) { if (t < s) red[t] += red[t + s]; __syncthreads(); }
  float var = red[0] / (float)D;
  out[o] = d0 * rsqrtf(var + 1e-5f) * g[t] + beta[t];
}

// newX = yxa[batch] + (yxm[batch]+1) * (num / (den + 1e-16))
__global__ void node_agg(const float* __restrict__ num, const float* __restrict__ den,
                         const float* __restrict__ yxa, const float* __restrict__ yxm,
                         const int* __restrict__ batchv, float* __restrict__ newX) {
  int i = blockIdx.x * 256 + threadIdx.x;
  if (i >= NN * DXc) return;
  int r = i >> 8, f = i & 255;
  int b = batchv[r];
  float agg = num[i] / (den[i] + 1e-16f);
  newX[i] = yxa[b * DXc + f] + (yxm[b * DXc + f] + 1.f) * agg;
}

// scatter-accumulate pooling stats (sum, sumsq, min, max, count)
__global__ void pool_scatter(const float* __restrict__ x, const int* __restrict__ batchv,
                             const int* __restrict__ ind, int rows, int D,
                             float* __restrict__ sum, float* __restrict__ sq,
                             unsigned* __restrict__ mn, unsigned* __restrict__ mx,
                             float* __restrict__ cnt) {
  long long i = (long long)blockIdx.x * 256 + threadIdx.x;
  if (i >= (long long)rows * D) return;
  int r = (int)(i / D), f = (int)(i - (long long)r * D);
  int b = batchv[ind ? ind[r] : r];
  float xv = x[i];
  atomicAdd(&sum[b * D + f], xv);
  atomicAdd(&sq[b * D + f], xv * xv);
  atomicMin(&mn[b * D + f], flipf(xv));
  atomicMax(&mx[b * D + f], flipf(xv));
  if (f == 0) atomicAdd(&cnt[b], 1.f);
}

// finalize [mean | min | max | std] concat per graph
__global__ void pool_finish(const float* __restrict__ sum, const float* __restrict__ sq,
                            const unsigned* __restrict__ mn, const unsigned* __restrict__ mx,
                            const float* __restrict__ cnt, float* __restrict__ outc, int D) {
  int i = blockIdx.x * 256 + threadIdx.x;
  if (i >= BG * D) return;
  int b = i / D, f = i - b * D;
  float n = cnt[b];
  float mean = sum[i] / fmaxf(n, 1.f);
  float var = (sq[i] - n * mean * mean) / fmaxf(n - 1.f, 1.f);
  float sd = sqrtf(fmaxf(var, 0.f));
  float lo = (n > 0.5f) ? unflipf(mn[i]) : 0.f;
  float hi = (n > 0.5f) ? unflipf(mx[i]) : 0.f;
  float* o = outc + (size_t)b * 4 * D;
  o[f] = mean; o[D + f] = lo; o[2 * D + f] = hi; o[3 * D + f] = sd;
}

// ---------------------------------------------------------------------------
extern "C" void kernel_launch(void* const* d_in, const int* in_sizes, int n_in,
                              void* d_out, int out_size, void* d_ws, size_t ws_size,
                              hipStream_t stream) {
  (void)in_sizes; (void)n_in; (void)out_size; (void)ws_size;
  const float* X      = (const float*)d_in[0];
  const int*   eidx   = (const int*)d_in[1];     // [2, M]
  const float* eattr  = (const float*)d_in[2];
  const float* y      = (const float*)d_in[3];
  const int*   batchv = (const int*)d_in[4];
  auto P = [&](int i) { return (const float*)d_in[i]; };
  // params: 5..44 linear (w,b) pairs in spec order, 45..56 norm (g,b) pairs
  const float *qw=P(5),*qb=P(6),*kw=P(7),*kb=P(8),*vw=P(9),*vb=P(10);
  const float *eaw=P(11),*eab=P(12),*emw=P(13),*emb=P(14);
  const float *yeaw=P(15),*yeab=P(16),*yemw=P(17),*yemb=P(18);
  const float *yxaw=P(19),*yxab=P(20),*yxmw=P(21),*yxmb=P(22);
  const float *xow=P(23),*xob=P(24),*eow=P(25),*eob=P(26);
  const float *yyw=P(27),*yyb=P(28),*xyw=P(29),*xyb=P(30),*eyw=P(31),*eyb=P(32);
  const float *lx1w=P(33),*lx1b=P(34),*lx2w=P(35),*lx2b=P(36);
  const float *le1w=P(37),*le1b=P(38),*le2w=P(39),*le2b=P(40);
  const float *ly1w=P(41),*ly1b=P(42),*ly2w=P(43),*ly2b=P(44);
  const float *nx1g=P(45),*nx1b=P(46),*nx2g=P(47),*nx2b=P(48);
  const float *ne1g=P(49),*ne1b=P(50),*ne2g=P(51),*ne2b=P(52);
  const float *ny1g=P(53),*ny1b=P(54),*ny2g=P(55),*ny2b=P(56);

  // ---- workspace layout (~0.9 GB fp32) ----
  float* ws = (float*)d_ws;
  size_t cur = 0;
  auto alloc = [&](size_t n) { float* p = ws + cur; cur += n; return p; };
  float* q    = alloc((size_t)NN * DXc);
  float* k    = alloc((size_t)NN * DXc);
  float* v    = alloc((size_t)NN * DXc);
  unsigned* mx = (unsigned*)alloc((size_t)NN * DXc);
  float* den  = alloc((size_t)NN * DXc);
  float* num  = alloc((size_t)NN * DXc);
  float* yea  = alloc((size_t)BG * DXc);
  float* yem  = alloc((size_t)BG * DXc);
  float* yxa  = alloc((size_t)BG * DXc);
  float* yxm  = alloc((size_t)BG * DXc);
  float* newedge = alloc((size_t)MEDG * DEc);   // later reused as FFN-E tmp
  float* newX = alloc((size_t)NN * DXc);        // later reused as FFN-X tmp
  float* newx = alloc((size_t)NN * DXc);
  float* Xr   = alloc((size_t)NN * DXc);
  float* Er   = alloc((size_t)MEDG * DEc);
  float* hX   = alloc((size_t)NN * FFXc);
  float* hE   = alloc((size_t)MEDG * FFEc);
  float* sumX = alloc((size_t)BG * DXc);
  float* sqX  = alloc((size_t)BG * DXc);
  unsigned* mnX = (unsigned*)alloc((size_t)BG * DXc);
  unsigned* mxX = (unsigned*)alloc((size_t)BG * DXc);
  float* cntX = alloc(BG);
  float* sumE = alloc((size_t)BG * DEc);
  float* sqE  = alloc((size_t)BG * DEc);
  unsigned* mnE = (unsigned*)alloc((size_t)BG * DEc);
  unsigned* mxE = (unsigned*)alloc((size_t)BG * DEc);
  float* cntE = alloc(BG);
  float* poolXc = alloc((size_t)BG * 4 * DXc);
  float* poolEc = alloc((size_t)BG * 4 * DEc);
  float* newy = alloc((size_t)BG * DYc);
  float* yr   = alloc((size_t)BG * DYc);
  float* hy   = alloc((size_t)BG * FFYc);
  float* tmpy = alloc((size_t)BG * DYc);

  float* outX = (float*)d_out;
  float* outE = outX + (size_t)NN * DXc;
  float* outY = outE + (size_t)MEDG * DEc;

  auto gemm = [&](const float* A, const float* W, const float* b, float* C,
                  int Mr, int K, int Nn, bool relu, bool acc) {
    dim3 g(Nn / 64, (Mr + 63) / 64);
    if (relu)      gemm_wmma<true,  false><<<g, 256, 0, stream>>>(A, W, b, C, Mr, K, Nn);
    else if (acc)  gemm_wmma<false, true ><<<g, 256, 0, stream>>>(A, W, b, C, Mr, K, Nn);
    else           gemm_wmma<false, false><<<g, 256, 0, stream>>>(A, W, b, C, Mr, K, Nn);
  };
  auto fillf = [&](float* p, float vv, size_t n) {
    fill_f32<<<dim3((unsigned)((n + 255) / 256)), 256, 0, stream>>>(p, vv, (int)n);
  };
  auto fillu = [&](unsigned* p, unsigned vv, size_t n) {
    fill_u32<<<dim3((unsigned)((n + 255) / 256)), 256, 0, stream>>>(p, vv, (int)n);
  };
  auto ln = [&](const float* a, const float* add, const float* g, const float* b,
                float* o, int rows, int D) {
    ln_kernel<<<dim3(rows), D, 0, stream>>>(a, add, g, b, o, D);
  };

  // ---- init scatter buffers ----
  fillf(den, 0.f, (size_t)NN * DXc);
  fillf(num, 0.f, (size_t)NN * DXc);
  fillu(mx, host_flip(-1e30f), (size_t)NN * DXc);
  fillf(sumX, 0.f, (size_t)BG * DXc);  fillf(sqX, 0.f, (size_t)BG * DXc);
  fillu(mnX, host_flip(1e30f), (size_t)BG * DXc);
  fillu(mxX, host_flip(-1e30f), (size_t)BG * DXc);
  fillf(cntX, 0.f, BG);
  fillf(sumE, 0.f, (size_t)BG * DEc);  fillf(sqE, 0.f, (size_t)BG * DEc);
  fillu(mnE, host_flip(1e30f), (size_t)BG * DEc);
  fillu(mxE, host_flip(-1e30f), (size_t)BG * DEc);
  fillf(cntE, 0.f, BG);

  // ---- q/k/v and y-conditioning tables (WMMA GEMMs) ----
  gemm(X, qw, qb, q, NN, DXc, DXc, false, false);
  gemm(X, kw, kb, k, NN, DXc, DXc, false, false);
  gemm(X, vw, vb, v, NN, DXc, DXc, false, false);
  gemm(y, yeaw, yeab, yea, BG, DYc, DXc, false, false);
  gemm(y, yemw, yemb, yem, BG, DYc, DXc, false, false);
  gemm(y, yxaw, yxab, yxa, BG, DYc, DXc, false, false);
  gemm(y, yxmw, yxmb, yxm, BG, DYc, DXc, false, false);

  // ---- fused edge passes ----
  dim3 eg((MEDG + 15) / 16);
  edge_kernel<0><<<eg, 256, 0, stream>>>(eidx, eattr, batchv, q, k, v,
      emw, emb, eaw, eab, eow, eob, yea, yem, mx, den, num, newedge);
  edge_kernel<1><<<eg, 256, 0, stream>>>(eidx, eattr, batchv, q, k, v,
      emw, emb, eaw, eab, eow, eob, yea, yem, mx, den, num, newedge);

  // ---- node aggregation + x_out ----
  node_agg<<<dim3((NN * DXc + 255) / 256), 256, 0, stream>>>(num, den, yxa, yxm,
                                                             batchv, newX);
  gemm(newX, xow, xob, newx, NN, DXc, DXc, false, false);

  // ---- residual + LN1 ----
  ln(X, newx, nx1g, nx1b, Xr, NN, DXc);
  ln(eattr, newedge, ne1g, ne1b, Er, MEDG, DEc);

  // ---- pooling + new_y ----
  pool_scatter<<<dim3((NN * DXc + 255) / 256), 256, 0, stream>>>(
      X, batchv, nullptr, NN, DXc, sumX, sqX, mnX, mxX, cntX);
  pool_scatter<<<dim3((unsigned)(((size_t)MEDG * DEc + 255) / 256)), 256, 0, stream>>>(
      eattr, batchv, eidx /* src row -> graph via batch[src] */, MEDG, DEc,
      sumE, sqE, mnE, mxE, cntE);
  pool_finish<<<dim3((BG * DXc + 255) / 256), 256, 0, stream>>>(sumX, sqX, mnX, mxX,
                                                                cntX, poolXc, DXc);
  pool_finish<<<dim3((BG * DEc + 255) / 256), 256, 0, stream>>>(sumE, sqE, mnE, mxE,
                                                                cntE, poolEc, DEc);
  gemm(y,      yyw, yyb, newy, BG, DYc,      DYc, false, false);
  gemm(poolXc, xyw, xyb, newy, BG, 4 * DXc,  DYc, false, true);
  gemm(poolEc, eyw, eyb, newy, BG, 4 * DEc,  DYc, false, true);
  ln(y, newy, ny1g, ny1b, yr, BG, DYc);

  // ---- FFN branches + LN2 -> outputs ----
  gemm(Xr, lx1w, lx1b, hX, NN, DXc, FFXc, true, false);
  gemm(hX, lx2w, lx2b, newX /* reuse as tmpX */, NN, FFXc, DXc, false, false);
  ln(Xr, newX, nx2g, nx2b, outX, NN, DXc);

  gemm(Er, le1w, le1b, hE, MEDG, DEc, FFEc, true, false);
  gemm(hE, le2w, le2b, newedge /* reuse as tmpE */, MEDG, FFEc, DEc, false, false);
  ln(Er, newedge, ne2g, ne2b, outE, MEDG, DEc);

  gemm(yr, ly1w, ly1b, hy, BG, DYc, FFYc, true, false);
  gemm(hy, ly2w, ly2b, tmpy, BG, FFYc, DYc, false, false);
  ln(yr, tmpy, ny2g, ny2b, outY, BG, DYc);
}
